// DRKGModel_50105088475140
// MI455X (gfx1250) — compile-verified, matrix operand order checked by reading, not run on hardware
//
#include <hip/hip_runtime.h>

#define N_C   100000
#define N_PC  10000
#define NEDGE 1000000
#define DIM   128
#define NL    3

typedef __attribute__((ext_vector_type(16))) __bf16 v16bf;
typedef __attribute__((ext_vector_type(8)))  float  v8f;

// ---------------- utility kernels ----------------

__global__ void drkg_zero_f32(float* __restrict__ p, long n) {
  long i4 = ((long)blockIdx.x * blockDim.x + threadIdx.x) * 4;
  if (i4 < n) *(float4*)(p + i4) = make_float4(0.f, 0.f, 0.f, 0.f);
}

// one wave (32 lanes) per edge: float4 gather of h[src], atomic scatter onto msg[dst]
__global__ void drkg_scatter_add(const float* __restrict__ h, const int* __restrict__ src,
                                 const int* __restrict__ dst, float* __restrict__ msg,
                                 float* __restrict__ deg, int nE) {
  int t = blockIdx.x * blockDim.x + threadIdx.x;
  int e = t >> 5;
  int lane = t & 31;
  if (e >= nE) return;
  int s = src[e], d = dst[e];
  float4 v = *(const float4*)(h + (size_t)s * DIM + lane * 4);
  float* m = msg + (size_t)d * DIM + lane * 4;
  atomicAdd(m + 0, v.x);
  atomicAdd(m + 1, v.y);
  atomicAdd(m + 2, v.z);
  atomicAdd(m + 3, v.w);
  if (lane == 0) atomicAdd(deg + d, 1.0f);
}

// Pack one 128x128 f32 weight matrix (optionally the sum of two) into
// fragment-ordered bf16: dst[((nt*4 + kc)*32 + lane)*16 + e], so a B fragment
// for (col-tile nt, K-chunk kc) is 32 contiguous bytes per lane -> one 32B
// vector load (2x global_load_b128, coalesced across the wave) in the GEMM.
__global__ void drkg_pack_w(const float* __restrict__ W, const float* __restrict__ W2,
                            __bf16* __restrict__ dst) {
  int t = blockIdx.x * blockDim.x + threadIdx.x;   // 0 .. 16383
  int e    = t & 15;
  int lane = (t >> 4) & 31;
  int kc   = (t >> 9) & 3;
  int nt   = t >> 11;
  int half = lane >> 4;
  int k0   = kc * 32 + half * 8;
  int K    = (e < 8) ? (k0 + e) : (k0 + 16 + (e - 8));
  int col  = nt * 16 + (lane & 15);
  float v = W[K * DIM + col];
  if (W2) v += W2[K * DIM + col];
  dst[t] = (__bf16)v;
}

// ---------------- fused GEMM (+mean scale) + bias + LayerNorm + ReLU ----------------
// out = LN( sum_im scale_im(A_im) @ W_im + bias ) [relu], per 16-row block.
// 256 threads = 8 waves; wave w owns output columns [16w, 16w+16).
// A tiles staged via LDS as bf16 (per-row 1/max(deg,1) fused), B fragments read
// from the packed bf16 buffer, v_wmma_f32_16x16x32_bf16 over 4 K-chunks,
// accumulators spilled to LDS for the cross-wave LayerNorm (16-lane shfl_xor).
__launch_bounds__(256)
__global__ void drkg_gemm_ln(const float* __restrict__ A0,
                             const float* __restrict__ A1, const float* __restrict__ dg1,
                             const float* __restrict__ A2, const float* __restrict__ dg2,
                             const __bf16* __restrict__ Wpack, int nIn,
                             const float* __restrict__ bias0, const float* __restrict__ bias1,
                             const float* __restrict__ gamma, const float* __restrict__ beta,
                             float* __restrict__ out, int relu) {
  __shared__ __bf16 aT[16 * DIM];   // 4 KB bf16 A tile
  __shared__ float  cT[16 * DIM];   // 8 KB f32 accum tile for LN

  const int tid  = threadIdx.x;
  const int wave = tid >> 5;
  const int lane = tid & 31;
  const int half = lane >> 4;       // ISA 16-bit A layout: lanes 0-15 vs 16-31
  const int mn   = lane & 15;
  const int row0 = blockIdx.x * 16;

  v8f acc = {0.f, 0.f, 0.f, 0.f, 0.f, 0.f, 0.f, 0.f};

  const float* As[3]  = {A0, A1, A2};
  const float* dgs[3] = {nullptr, dg1, dg2};

  for (int im = 0; im < nIn; ++im) {
    const float* A = As[im];
    __syncthreads();  // previous iteration's aT reads complete
    {
      // cooperative A-tile load: 16 rows x 128 cols, 8 floats/thread,
      // fused mean division (scale = 1/max(deg,1)) and f32->bf16 convert
      const int r  = tid >> 4;
      const int c0 = (tid & 15) * 8;
      float s = 1.0f;
      if (dgs[im]) s = 1.0f / fmaxf(dgs[im][row0 + r], 1.0f);
      const float* ap = A + (size_t)(row0 + r) * DIM + c0;
      float4 u0 = *(const float4*)(ap);
      float4 u1 = *(const float4*)(ap + 4);
      __bf16* at = aT + r * DIM + c0;
      at[0] = (__bf16)(u0.x * s); at[1] = (__bf16)(u0.y * s);
      at[2] = (__bf16)(u0.z * s); at[3] = (__bf16)(u0.w * s);
      at[4] = (__bf16)(u1.x * s); at[5] = (__bf16)(u1.y * s);
      at[6] = (__bf16)(u1.z * s); at[7] = (__bf16)(u1.w * s);
    }
    __syncthreads();
    // packed B fragments for this (im, wave): frag index = im*1024 + (wave*4+kc)*32 + lane
    const __bf16* wb = Wpack + ((size_t)im * 1024 + (size_t)wave * 4 * 32 + lane) * 16;
#pragma unroll
    for (int kc = 0; kc < 4; ++kc) {
      const int k0 = kc * 32 + half * 8;
      v16bf a, b;
      b = *(const v16bf*)(wb + (size_t)kc * 32 * 16);   // 32B contiguous per lane
#pragma unroll
      for (int e = 0; e < 8; ++e) {
        // A frag (16-bit 16x32 layout): elems 0-7 -> K=k0+e, elems 8-15 -> K=k0+16+e
        a[e]     = aT[mn * DIM + k0 + e];
        a[e + 8] = aT[mn * DIM + k0 + 16 + e];
      }
      acc = __builtin_amdgcn_wmma_f32_16x16x32_bf16(false, a, false, b,
                                                    (short)0, acc, false, false);
    }
  }

  // D-matrix layout: VGPR j -> row (j + 8*half), col = wave*16 + (lane&15)
  __syncthreads();
#pragma unroll
  for (int j = 0; j < 8; ++j)
    cT[(j + 8 * half) * DIM + wave * 16 + mn] = acc[j];
  __syncthreads();

  // LayerNorm epilogue: 16 contiguous threads per row (aligned 16-lane group in wave32)
  {
    const int r  = tid >> 4;
    const int c0 = (tid & 15) * 8;
    float v[8];
    float sum = 0.f, sq = 0.f;
#pragma unroll
    for (int j = 0; j < 8; ++j) {
      const int c = c0 + j;
      float x = cT[r * DIM + c] + bias0[c];
      if (bias1) x += bias1[c];
      v[j] = x;
      sum += x;
      sq  += x * x;
    }
#pragma unroll
    for (int off = 8; off >= 1; off >>= 1) {
      sum += __shfl_xor(sum, off, 16);
      sq  += __shfl_xor(sq,  off, 16);
    }
    const float mean = sum * (1.0f / DIM);
    const float var  = sq * (1.0f / DIM) - mean * mean;
    const float rstd = rsqrtf(var + 1e-5f);
    float* op = out + (size_t)(row0 + r) * DIM;
#pragma unroll
    for (int j = 0; j < 8; ++j) {
      const int c = c0 + j;
      float y = (v[j] - mean) * rstd * gamma[c] + beta[c];
      if (relu) y = fmaxf(y, 0.f);
      op[c] = y;
    }
  }
}

// ---------------- driver ----------------

extern "C" void kernel_launch(void* const* d_in, const int* in_sizes, int n_in,
                              void* d_out, int out_size, void* d_ws, size_t ws_size,
                              hipStream_t stream) {
  (void)in_sizes; (void)n_in; (void)out_size; (void)ws_size;
  const float* x_c    = (const float*)d_in[0];
  const float* x_pc   = (const float*)d_in[1];
  const int*   src0   = (const int*)d_in[2];
  const int*   dst0   = (const int*)d_in[3];
  const int*   src1   = (const int*)d_in[4];
  const int*   dst1   = (const int*)d_in[5];
  const float* Wself  = (const float*)d_in[6];   // [L,2,D,D]
  const float* Wneigh = (const float*)d_in[7];   // [L,2,D,D]
  const float* bneigh = (const float*)d_in[8];   // [L,2,D]
  const float* linW   = (const float*)d_in[9];   // [L,D,D]
  const float* linb   = (const float*)d_in[10];  // [L,D]
  const float* gamma  = (const float*)d_in[11];  // [L,D]
  const float* beta   = (const float*)d_in[12];  // [L,D]
  float* out = (float*)d_out;

  float* ws = (float*)d_ws;
  size_t o = 0;
  float* hA   = ws + o; o += (size_t)N_C * DIM;
  float* hB   = ws + o; o += (size_t)N_C * DIM;
  float* pcA  = ws + o; o += (size_t)N_PC * DIM;
  float* pcB  = ws + o; o += (size_t)N_PC * DIM;
  float* msg0 = ws + o; o += (size_t)N_C * DIM;   // msg0,msg1,deg0,deg1 contiguous
  float* msg1 = ws + o; o += (size_t)N_C * DIM;
  float* deg0 = ws + o; o += N_C;
  float* deg1 = ws + o; o += N_C;
  __bf16* WpC  = (__bf16*)(ws + o); o += 3 * DIM * DIM / 2;  // 3 packed matrices (bf16)
  __bf16* WpPC = (__bf16*)(ws + o); o += DIM * DIM / 2;      // 1 packed matrix (bf16)

  const long zeroN = 2L * N_C * DIM + 2L * N_C;   // spans msg0..deg1, multiple of 4
  const int  zeroBlocks = (int)((zeroN / 4 + 255) / 256);
  const int  scatBlocks = (int)(((long)NEDGE * 32 + 255) / 256);
  const int  packBlocks = (DIM * DIM) / 256;      // 64

  for (int l = 0; l < NL; ++l) {
    const float* hin  = (l == 0) ? x_c  : ((l == 1) ? hA : hB);
    float*       hout = (l == 2) ? (out + (size_t)N_PC * DIM) : ((l == 0) ? hA : hB);
    const float* pin  = (l == 0) ? x_pc : ((l == 1) ? pcA : pcB);
    float*       pout = (l == 2) ? out : ((l == 0) ? pcA : pcB);

    drkg_zero_f32<<<zeroBlocks, 256, 0, stream>>>(msg0, zeroN);
    drkg_scatter_add<<<scatBlocks, 256, 0, stream>>>(hin, src0, dst0, msg0, deg0, NEDGE);
    drkg_scatter_add<<<scatBlocks, 256, 0, stream>>>(hin, src1, dst1, msg1, deg1, NEDGE);

    // pack weights (bf16, fragment-ordered); Wsum = Ws0+Ws1 fused into pack
    drkg_pack_w<<<packBlocks, 256, 0, stream>>>(
        Wself + (size_t)(l * 2) * DIM * DIM,
        Wself + (size_t)(l * 2 + 1) * DIM * DIM, WpC);
    drkg_pack_w<<<packBlocks, 256, 0, stream>>>(
        Wneigh + (size_t)(l * 2) * DIM * DIM, nullptr, WpC + DIM * DIM);
    drkg_pack_w<<<packBlocks, 256, 0, stream>>>(
        Wneigh + (size_t)(l * 2 + 1) * DIM * DIM, nullptr, WpC + 2 * DIM * DIM);
    drkg_pack_w<<<packBlocks, 256, 0, stream>>>(
        linW + (size_t)l * DIM * DIM, nullptr, WpPC);

    // compound path: h@Wsum + mean0@Wn0 + mean1@Wn1 + b0 + b1 -> LN -> relu?
    drkg_gemm_ln<<<N_C / 16, 256, 0, stream>>>(
        hin, msg0, deg0, msg1, deg1,
        WpC, 3,
        bneigh + (size_t)(l * 2) * DIM,
        bneigh + (size_t)(l * 2 + 1) * DIM,
        gamma + (size_t)l * DIM, beta + (size_t)l * DIM,
        hout, (l < NL - 1) ? 1 : 0);

    // PC path: pc@linW + linb -> LN -> relu?
    drkg_gemm_ln<<<N_PC / 16, 256, 0, stream>>>(
        pin, nullptr, nullptr, nullptr, nullptr,
        WpPC, 1,
        linb + (size_t)l * DIM, nullptr,
        gamma + (size_t)l * DIM, beta + (size_t)l * DIM,
        pout, (l < NL - 1) ? 1 : 0);
  }
}
